// GraphSAGE_79405355368448
// MI455X (gfx1250) — matrix operand order repaired and checked by reading, not compiled, not run on hardware
//
#include <hip/hip_runtime.h>
#include <hip/hip_bf16.h>

typedef __attribute__((ext_vector_type(16))) _Float16 v16h;
typedef __attribute__((ext_vector_type(8)))  _Float16 v8h;
typedef __attribute__((ext_vector_type(8)))  float    v8f;

#define NN 100000
#define FD 128   // input feature dim for both layers

// ---------------- utility: zero fill ----------------
__global__ void zero_f32(float* __restrict__ p, int n) {
    int i = blockIdx.x * blockDim.x + threadIdx.x;
    if (i < n) p[i] = 0.0f;
}

// ---------------- degree accumulation ----------------
__global__ void degree_kernel(const long long* __restrict__ dst,
                              float* __restrict__ deg, int E) {
    int e = blockIdx.x * blockDim.x + threadIdx.x;
    if (e < E) atomicAdd(&deg[(int)dst[e]], 1.0f);
}

__global__ void invdeg_kernel(float* __restrict__ deg, int n) {
    int i = blockIdx.x * blockDim.x + threadIdx.x;
    if (i < n) deg[i] = 1.0f / fmaxf(deg[i], 1.0f);
}

// ---------------- edge scatter: agg[dst] += feat[src] ----------------
// one wave per edge; each lane moves 4 consecutive floats (32*4 = 128)
__global__ void scatter_kernel(const float* __restrict__ feat,
                               const long long* __restrict__ src,
                               const long long* __restrict__ dst,
                               float* __restrict__ agg, int E) {
    int gid  = blockIdx.x * blockDim.x + threadIdx.x;
    int e    = gid >> 5;
    int lane = gid & 31;
    if (e >= E) return;
    int s = (int)src[e];
    int d = (int)dst[e];
    const float4 v = *reinterpret_cast<const float4*>(feat + (size_t)s * FD + lane * 4);
    float* a = agg + (size_t)d * FD + lane * 4;
    atomicAdd(a + 0, v.x);
    atomicAdd(a + 1, v.y);
    atomicAdd(a + 2, v.z);
    atomicAdd(a + 3, v.w);
}

// ---------------- fused SAGE GEMM (WMMA f16 -> f32) ----------------
// out[m, n] = relu_opt( sum_k (agg[m,k]*invdeg[m]) * Wl[n,k]
//                     + sum_k  x[m,k]              * Wr[n,k]  + bias[n] )
// block: 16 node rows; blockDim = (32, N/16): one wave per 16-column tile.
__global__ void __launch_bounds__(256)
sage_gemm_wmma(const float* __restrict__ agg,
               const float* __restrict__ invdeg,
               const float* __restrict__ x,
               const float* __restrict__ Wl,
               const float* __restrict__ Wr,
               const float* __restrict__ bias,
               float* __restrict__ out,
               int M, int N, int relu) {
    __shared__ alignas(16) _Float16 As[16][FD];
    __shared__ alignas(16) _Float16 Xs[16][FD];

    const int lane     = threadIdx.x;       // 0..31
    const int wv       = threadIdx.y;       // column-tile index
    const int tid      = wv * 32 + lane;
    const int nthreads = blockDim.y * 32;
    const int row0     = blockIdx.x * 16;

    // Stage 16x128 tiles of (agg*invdeg) and x into LDS as f16.
    for (int idx = tid; idx < 16 * FD; idx += nthreads) {
        int r = idx >> 7;        // /128
        int k = idx & 127;
        int node = row0 + r;
        float a = 0.0f, xv = 0.0f;
        if (node < M) {
            a  = agg[(size_t)node * FD + k] * invdeg[node];
            xv = x  [(size_t)node * FD + k];
        }
        As[r][k] = (_Float16)a;
        Xs[r][k] = (_Float16)xv;
    }
    __syncthreads();

    const int n0   = wv * 16;
    const int ncol = n0 + (lane & 15);
    const int kb16 = (lane < 16) ? 0 : 16;   // B-frag K base within 32-step
    const int kb8  = (lane < 16) ? 0 : 16;   // A-frag K base within 32-step
    const int mrow = lane & 15;

    // Preload B fragments for all 4 K-steps (Wl and Wr), K-contiguous.
    v16h bl[4], br[4];
#pragma unroll
    for (int s = 0; s < 4; ++s) {
        const float4* pl = reinterpret_cast<const float4*>(Wl + (size_t)ncol * FD + 32 * s + kb16);
        const float4* pr = reinterpret_cast<const float4*>(Wr + (size_t)ncol * FD + 32 * s + kb16);
#pragma unroll
        for (int i = 0; i < 4; ++i) {
            float4 ql = pl[i];
            float4 qr = pr[i];
            bl[s][4 * i + 0] = (_Float16)ql.x; bl[s][4 * i + 1] = (_Float16)ql.y;
            bl[s][4 * i + 2] = (_Float16)ql.z; bl[s][4 * i + 3] = (_Float16)ql.w;
            br[s][4 * i + 0] = (_Float16)qr.x; br[s][4 * i + 1] = (_Float16)qr.y;
            br[s][4 * i + 2] = (_Float16)qr.z; br[s][4 * i + 3] = (_Float16)qr.w;
        }
    }

    v8f acc = {};
#pragma unroll
    for (int s = 0; s < 4; ++s) {
        // A fragment, K-contiguous to match B: halves 0..7 -> K = kb8..kb8+7,
        //                            halves 8..15 -> K = kb8+8..kb8+15
        v8h a0 = *reinterpret_cast<const v8h*>(&As[mrow][32 * s + kb8]);
        v8h a1 = *reinterpret_cast<const v8h*>(&As[mrow][32 * s + kb8 + 8]);
        v8h x0 = *reinterpret_cast<const v8h*>(&Xs[mrow][32 * s + kb8]);
        v8h x1 = *reinterpret_cast<const v8h*>(&Xs[mrow][32 * s + kb8 + 8]);
        v16h afrag, xfrag;
#pragma unroll
        for (int h = 0; h < 8; ++h) {
            afrag[h] = a0[h]; afrag[h + 8] = a1[h];
            xfrag[h] = x0[h]; xfrag[h + 8] = x1[h];
        }
        acc = __builtin_amdgcn_wmma_f32_16x16x32_f16(false, afrag, false, bl[s],
                                                     (short)0, acc, false, false);
        acc = __builtin_amdgcn_wmma_f32_16x16x32_f16(false, xfrag, false, br[s],
                                                     (short)0, acc, false, false);
    }

    // Epilogue: D layout lane L -> N = L%16; VGPR r -> M = r + (L<16 ? 0 : 8)
    const float bv   = bias[ncol];
    const int mbase  = (lane < 16) ? 0 : 8;
#pragma unroll
    for (int r = 0; r < 8; ++r) {
        int node = row0 + mbase + r;
        if (node < M) {
            float v = acc[r] + bv;
            if (relu) v = fmaxf(v, 0.0f);
            out[(size_t)node * N + ncol] = v;
        }
    }
}

// ---------------- row-wise log-softmax over 64 classes ----------------
// one wave per node; each lane owns 2 consecutive columns (float2).
__global__ void log_softmax_kernel(float* __restrict__ out, int M) {
    int gid  = blockIdx.x * blockDim.x + threadIdx.x;
    int node = gid >> 5;
    int lane = gid & 31;
    if (node >= M) return;
    float2* row = reinterpret_cast<float2*>(out + (size_t)node * 64);
    float2 v = row[lane];
    float m = fmaxf(v.x, v.y);
#pragma unroll
    for (int off = 16; off > 0; off >>= 1) m = fmaxf(m, __shfl_xor(m, off, 32));
    float s = __expf(v.x - m) + __expf(v.y - m);
#pragma unroll
    for (int off = 16; off > 0; off >>= 1) s += __shfl_xor(s, off, 32);
    float lse = m + __logf(s);
    float2 r;
    r.x = v.x - lse;
    r.y = v.y - lse;
    row[lane] = r;
}

// ---------------- launcher ----------------
extern "C" void kernel_launch(void* const* d_in, const int* in_sizes, int n_in,
                              void* d_out, int out_size, void* d_ws, size_t ws_size,
                              hipStream_t stream) {
    const float*     x    = (const float*)d_in[0];
    const long long* edge = (const long long*)d_in[1];   // int64 per reference
    const float*     W1l  = (const float*)d_in[2];
    const float*     b1   = (const float*)d_in[3];
    const float*     W1r  = (const float*)d_in[4];
    const float*     W2l  = (const float*)d_in[5];
    const float*     b2   = (const float*)d_in[6];
    const float*     W2r  = (const float*)d_in[7];
    float*           out  = (float*)d_out;

    const int M = NN;
    const int E = in_sizes[1] / 2;
    const long long* src = edge;
    const long long* dst = edge + E;

    // workspace: deg/invdeg [M] | agg [M*128] | h [M*128]
    float* deg = (float*)d_ws;
    float* agg = deg + ((M + 255) & ~255);
    float* h   = agg + (size_t)M * FD;

    const int nAgg = M * FD;

    // degrees (shared by both layers)
    zero_f32<<<(M + 255) / 256, 256, 0, stream>>>(deg, M);
    degree_kernel<<<(E + 255) / 256, 256, 0, stream>>>(dst, deg, E);
    invdeg_kernel<<<(M + 255) / 256, 256, 0, stream>>>(deg, M);

    // ----- layer 1 -----
    zero_f32<<<(nAgg + 255) / 256, 256, 0, stream>>>(agg, nAgg);
    scatter_kernel<<<((size_t)E * 32 + 255) / 256, 256, 0, stream>>>(x, src, dst, agg, E);
    sage_gemm_wmma<<<dim3((M + 15) / 16), dim3(32, 8), 0, stream>>>(
        agg, deg, x, W1l, W1r, b1, h, M, 128, /*relu=*/1);

    // ----- layer 2 -----
    zero_f32<<<(nAgg + 255) / 256, 256, 0, stream>>>(agg, nAgg);
    scatter_kernel<<<((size_t)E * 32 + 255) / 256, 256, 0, stream>>>(h, src, dst, agg, E);
    sage_gemm_wmma<<<dim3((M + 15) / 16), dim3(32, 4), 0, stream>>>(
        agg, deg, h, W2l, W2r, b2, out, M, 64, /*relu=*/0);

    // log-softmax in place on d_out
    log_softmax_kernel<<<((size_t)M * 32 + 255) / 256, 256, 0, stream>>>(out, M);
}
